// VisionTemporalAttention_43585328119964
// MI455X (gfx1250) — matrix operand order repaired and checked by reading, not compiled, not run on hardware
//
#include <hip/hip_runtime.h>
#include <hip/hip_bf16.h>
#include <math.h>

typedef _Float16 f16;
typedef __attribute__((ext_vector_type(16))) _Float16 v16h;
typedef __attribute__((ext_vector_type(8)))  float    v8f;
typedef __attribute__((ext_vector_type(4)))  int      v4i;
typedef __attribute__((address_space(1))) v4i v4i_g;  // global int4
typedef __attribute__((address_space(3))) v4i v4i_l;  // LDS int4

#define B_     4
#define N_     4096
#define DIM_   1024
#define H_     4
#define DH_    64
#define INNER_ 256
#define ROT_   32

#if __has_builtin(__builtin_amdgcn_global_load_async_to_lds_b128)
#define ASYNC_LDS 1
#else
#define ASYNC_LDS 0
#endif

// ---------------------------------------------------------------------------
// WMMA helpers (CDNA5 V_WMMA_F32_16X16X32_F16, wave32)
// ---------------------------------------------------------------------------
__device__ __forceinline__ v8f wmma_f16(v16h a, v16h b, v8f c) {
  // (neg_a, A, neg_b, B, c_mod, C, reuse_a, reuse_b)
  return __builtin_amdgcn_wmma_f32_16x16x32_f16(false, a, false, b, (short)0, c,
                                                false, false);
}

// A fragment: 16x32 f16 (MxK), source row-major, p0 pre-offset to (m0,k0).
// ISA layout: lane L holds row M=L&15; element e -> K=(e&7)+8*hi+16*(e>>3).
__device__ __forceinline__ v16h load_a(const f16* __restrict__ p0, int ldm,
                                       int lane) {
  const int hi = (lane >> 4) & 1;
  const f16* p = p0 + (size_t)(lane & 15) * ldm + 8 * hi;
  v16h a;
#pragma unroll
  for (int e = 0; e < 8; ++e) a[e] = p[e];
#pragma unroll
  for (int e = 0; e < 8; ++e) a[8 + e] = p[16 + e];
  return a;
}

// B fragment: 32x16 f16 (KxN), source is "Bt" with Bt[n][k] row-major
// (row stride ldk). ISA layout: lane L holds col N=L&15; element e -> K=(e&7)+8*hi+16*(e>>3).
// p0 pre-offset to (n0,k0): two 8-half reads per lane.
__device__ __forceinline__ v16h load_bt(const f16* __restrict__ p0, int ldk,
                                        int lane) {
  const int hi = (lane >> 4) & 1;
  const f16* p = p0 + (size_t)(lane & 15) * ldk + 8 * hi;
  v16h b;
#pragma unroll
  for (int e = 0; e < 8; ++e) b[e] = p[e];
#pragma unroll
  for (int e = 0; e < 8; ++e) b[8 + e] = p[16 + e];
  return b;
}

__device__ __forceinline__ float rmax16(float v) {
  v = fmaxf(v, __shfl_xor(v, 8, 32));
  v = fmaxf(v, __shfl_xor(v, 4, 32));
  v = fmaxf(v, __shfl_xor(v, 2, 32));
  v = fmaxf(v, __shfl_xor(v, 1, 32));
  return v;
}
__device__ __forceinline__ float rsum16(float v) {
  v += __shfl_xor(v, 8, 32);
  v += __shfl_xor(v, 4, 32);
  v += __shfl_xor(v, 2, 32);
  v += __shfl_xor(v, 1, 32);
  return v;
}

// 16-byte global -> LDS copy (async on CDNA5; falls back to reg-staged)
__device__ __forceinline__ void cp16B(f16* l, const f16* g) {
#if ASYNC_LDS
  __builtin_amdgcn_global_load_async_to_lds_b128((v4i_g*)(g), (v4i_l*)(l), 0, 0);
#else
  *(uint4*)l = *(const uint4*)g;
#endif
}

#if ASYNC_LDS
#define WAIT_ASYNC_LE4() asm volatile("s_wait_asynccnt 4" ::: "memory")
#define WAIT_ASYNC_0()   asm volatile("s_wait_asynccnt 0" ::: "memory")
#else
#define WAIT_ASYNC_LE4()
#define WAIT_ASYNC_0()
#endif

// ---------------------------------------------------------------------------
// k0: LayerNorm (fp32 in) -> xn (f16)
// ---------------------------------------------------------------------------
__global__ __launch_bounds__(256) void k0_ln(const float* __restrict__ x,
                                             const float* __restrict__ w,
                                             const float* __restrict__ bb,
                                             f16* __restrict__ xn) {
  __shared__ float s1[256], s2[256];
  const int row = blockIdx.x;
  const int tid = threadIdx.x;
  const float* xr = x + (size_t)row * DIM_;
  float v[4];
  float sum = 0.f, sq = 0.f;
#pragma unroll
  for (int i = 0; i < 4; ++i) {
    v[i] = xr[tid + 256 * i];
    sum += v[i];
    sq += v[i] * v[i];
  }
  s1[tid] = sum;
  s2[tid] = sq;
  __syncthreads();
  for (int st = 128; st > 0; st >>= 1) {
    if (tid < st) {
      s1[tid] += s1[tid + st];
      s2[tid] += s2[tid + st];
    }
    __syncthreads();
  }
  const float mu = s1[0] * (1.0f / DIM_);
  const float var = s2[0] * (1.0f / DIM_) - mu * mu;
  const float rstd = rsqrtf(var + 1e-5f);
  f16* xo = xn + (size_t)row * DIM_;
#pragma unroll
  for (int i = 0; i < 4; ++i) {
    const int ci = tid + 256 * i;
    xo[ci] = (f16)((v[i] - mu) * rstd * w[ci] + bb[ci]);
  }
}

// Cast+transpose weights to f16 "Bt" layout (out col major -> rows)
__global__ void k0_twqkv(const float* __restrict__ w, f16* __restrict__ wT) {
  int idx = blockIdx.x * 256 + threadIdx.x;
  if (idx < DIM_ * 3 * INNER_) {
    int kk = idx / (3 * INNER_);
    int n = idx % (3 * INNER_);
    wT[(size_t)n * DIM_ + kk] = (f16)w[idx];
  }
}
__global__ void k0_twout(const float* __restrict__ w, f16* __restrict__ wT) {
  int idx = blockIdx.x * 256 + threadIdx.x;
  if (idx < INNER_ * DIM_) {
    int kk = idx / DIM_;
    int n = idx % DIM_;
    wT[(size_t)n * INNER_ + kk] = (f16)w[idx];
  }
}

// ---------------------------------------------------------------------------
// k1: QKV GEMM (WMMA) + bias + RoPE + q-scale; writes q,k [B,H,N,DH] and
//     v transposed [B,H,DH,N], all f16. One wave = 16 rows x 32 cols slab.
// ---------------------------------------------------------------------------
__global__ __launch_bounds__(128) void k1_qkv(
    const f16* __restrict__ xn, const f16* __restrict__ wT,
    const float* __restrict__ bias, const float* __restrict__ rope,
    f16* __restrict__ q, f16* __restrict__ k, f16* __restrict__ vT) {
  const int wave = threadIdx.x >> 5;
  const int lane = threadIdx.x & 31;
  const int hi = lane >> 4;
  const int c = lane & 15;
  const int slab = blockIdx.y * 4 + wave;  // 0..23
  const int mt = blockIdx.x;               // 0..1023
  const int col0 = slab * 32;

  v8f c0 = {}, c1 = {};
  const f16* arow = xn + (size_t)mt * 16 * DIM_;
  const f16* b0p = wT + (size_t)col0 * DIM_;
  const f16* b1p = wT + (size_t)(col0 + 16) * DIM_;
  for (int kk = 0; kk < DIM_; kk += 32) {
    v16h a = load_a(arow + kk, DIM_, lane);
    c0 = wmma_f16(a, load_bt(b0p + kk, DIM_, lane), c0);
    c1 = wmma_f16(a, load_bt(b1p + kk, DIM_, lane), c1);
  }
  const float bias0 = bias[col0 + c];
  const float bias1 = bias[col0 + 16 + c];
  const int region = col0 >> 8;  // 0:q 1:k 2:v
  const int rel = col0 & 255;
  const int h = rel >> 6;
  const int d0 = rel & 63;  // 0 or 32 (slabs are 32-aligned)
  const bool do_rope = (region < 2) && (d0 == 0);

#pragma unroll
  for (int r = 0; r < 8; ++r) {
    const int row = mt * 16 + r + 8 * hi;
    const int b_ = row >> 12;      // row / N_
    const int n = row & (N_ - 1);  // row % N_
    float t0 = c0[r] + bias0;
    float t1 = c1[r] + bias1;
    if (do_rope) {
      // pair (i, i+16) lives in (frag0, frag1) of the SAME lane/reg
      const float p0 = rope[n * ROT_ + c];
      const float p1 = rope[n * ROT_ + 16 + c];
      const float r0 = t0 * __cosf(p0) - t1 * __sinf(p0);
      const float r1 = t1 * __cosf(p1) + t0 * __sinf(p1);
      t0 = r0;
      t1 = r1;
    }
    if (region == 0) {
      t0 *= 0.125f;  // DH^-0.5 folded into q
      t1 *= 0.125f;
      f16* qp = q + ((size_t)(b_ * H_ + h) * N_ + n) * DH_ + d0;
      qp[c] = (f16)t0;
      qp[16 + c] = (f16)t1;
    } else if (region == 1) {
      f16* kp = k + ((size_t)(b_ * H_ + h) * N_ + n) * DH_ + d0;
      kp[c] = (f16)t0;
      kp[16 + c] = (f16)t1;
    } else {
      f16* vp = vT + (size_t)(b_ * H_ + h) * DH_ * N_;
      vp[(size_t)(d0 + c) * N_ + n] = (f16)t0;
      vp[(size_t)(d0 + 16 + c) * N_ + n] = (f16)t1;
    }
  }
}

// ---------------------------------------------------------------------------
// k2: flash attention. Block = 4 waves = 64 q rows of one (b,h).
// K/V tiles staged once per block in LDS via async-to-LDS, double buffered.
// ---------------------------------------------------------------------------
__global__ __launch_bounds__(128) void k2_attn(const f16* __restrict__ q,
                                               const f16* __restrict__ k,
                                               const f16* __restrict__ vT,
                                               const unsigned char* __restrict__ mask,
                                               f16* __restrict__ O) {
  __shared__ __align__(16) f16 lsK[2][32 * DH_];  // [buf][key][feat]  4KB/buf
  __shared__ __align__(16) f16 lsV[2][DH_ * 32];  // [buf][feat][key]  4KB/buf
  __shared__ f16 lsP[4][16 * 34];                 // per-wave P tile

  const int tid = threadIdx.x;
  const int wave = tid >> 5;
  const int lane = tid & 31;
  const int hi = lane >> 4;
  const int c = lane & 15;
  const int bh = blockIdx.x >> 6;  // 0..15
  const int b = bh >> 2;
  const int h = bh & (H_ - 1);
  const int q0 = (blockIdx.x & 63) * 64 + wave * 16;

  const f16* qb = q + (size_t)bh * N_ * DH_ + (size_t)q0 * DH_;
  const f16* kb = k + (size_t)bh * N_ * DH_;
  const f16* vb = vT + (size_t)bh * DH_ * N_;
  const unsigned char* mk = mask + (size_t)b * N_;

  const v16h aq0 = load_a(qb + 0, DH_, lane);   // K = 0..31
  const v16h aq1 = load_a(qb + 32, DH_, lane);  // K = 32..63

  float m_[8], l_[8];
  v8f acc0 = {}, acc1 = {}, acc2 = {}, acc3 = {};
#pragma unroll
  for (int r = 0; r < 8; ++r) {
    m_[r] = -3.0e38f;
    l_[r] = 0.f;
  }

  // cooperative tile fetch: 4 x 16B per thread (2 for K, 2 for V)
  auto fetch_tile = [&](int buf, int j0) {
#pragma unroll
    for (int p = 0; p < 2; ++p) {
      const int chunk = tid + 128 * p;  // 0..255, 16B each
      // K tile: keys j0..j0+31 row-major -> contiguous 4KB
      cp16B(&lsK[buf][chunk * 8], kb + (size_t)j0 * DH_ + chunk * 8);
      // V tile: 64 feature rows x 32 keys (64B per row, strided source)
      const int row = chunk >> 2;
      const int off = (chunk & 3) * 8;
      cp16B(&lsV[buf][row * 32 + off], vb + (size_t)row * N_ + j0 + off);
    }
  };

  fetch_tile(0, 0);  // prologue

  for (int it = 0; it < N_ / 32; ++it) {
    const int j0 = it * 32;
    const int buf = it & 1;
    const bool has_next = (it + 1) < (N_ / 32);
    if (has_next) {
      fetch_tile(buf ^ 1, j0 + 32);  // overlap copy with compute
      WAIT_ASYNC_LE4();              // current tile's 4 asyncs done
    } else {
      WAIT_ASYNC_0();
    }
    __syncthreads();  // all waves' copies for `buf` visible

    const f16* lk = &lsK[buf][0];
    const f16* lv = &lsV[buf][0];

    // scores: sim[16 x 32] via 4 WMMA (two 16-col tiles, K=64 in two steps)
    v8f s0 = {}, s1 = {};
    s0 = wmma_f16(aq0, load_bt(lk + 0 * DH_ + 0, DH_, lane), s0);
    s0 = wmma_f16(aq1, load_bt(lk + 0 * DH_ + 32, DH_, lane), s0);
    s1 = wmma_f16(aq0, load_bt(lk + 16 * DH_ + 0, DH_, lane), s1);
    s1 = wmma_f16(aq1, load_bt(lk + 16 * DH_ + 32, DH_, lane), s1);

    const bool ok0 = mk[j0 + c] != 0;
    const bool ok1 = mk[j0 + 16 + c] != 0;
    f16* lp = &lsP[wave][0];
#pragma unroll
    for (int r = 0; r < 8; ++r) {
      float v0 = ok0 ? s0[r] : -3.0e38f;
      float v1 = ok1 ? s1[r] : -3.0e38f;
      float tm = rmax16(fmaxf(v0, v1));
      float nm = fmaxf(m_[r], tm);
      float corr = __expf(m_[r] - nm);
      float p0 = __expf(v0 - nm);
      float p1 = __expf(v1 - nm);
      float rs = rsum16(p0 + p1);
      l_[r] = l_[r] * corr + rs;
      m_[r] = nm;
      acc0[r] *= corr;
      acc1[r] *= corr;
      acc2[r] *= corr;
      acc3[r] *= corr;
      lp[(r + 8 * hi) * 34 + c] = (f16)p0;
      lp[(r + 8 * hi) * 34 + 16 + c] = (f16)p1;
    }
    asm volatile("s_wait_dscnt 0" ::: "memory");  // C-layout -> A-layout
    v16h pa;
    {
      const f16* sp = &lsP[wave][(lane & 15) * 34 + 8 * hi];
#pragma unroll
      for (int e = 0; e < 8; ++e) pa[e] = sp[e];
#pragma unroll
      for (int e = 0; e < 8; ++e) pa[8 + e] = sp[16 + e];
    }
    // out += P(16x32) @ V(32x64): 4 WMMA, lsV rows are features
    acc0 = wmma_f16(pa, load_bt(lv + 0 * 32, 32, lane), acc0);
    acc1 = wmma_f16(pa, load_bt(lv + 16 * 32, 32, lane), acc1);
    acc2 = wmma_f16(pa, load_bt(lv + 32 * 32, 32, lane), acc2);
    acc3 = wmma_f16(pa, load_bt(lv + 48 * 32, 32, lane), acc3);

    __syncthreads();  // all reads of `buf` done before it is refilled
  }

#pragma unroll
  for (int r = 0; r < 8; ++r) {
    const float inv = (l_[r] > 0.f) ? (1.0f / l_[r]) : 0.f;
    const int nq = q0 + r + 8 * hi;
    f16* op = O + (size_t)(b * N_ + nq) * INNER_ + h * DH_;
    op[0 + c] = (f16)(acc0[r] * inv);
    op[16 + c] = (f16)(acc1[r] * inv);
    op[32 + c] = (f16)(acc2[r] * inv);
    op[48 + c] = (f16)(acc3[r] * inv);
  }
}

// ---------------------------------------------------------------------------
// k3: out = O(f16, 16384x256) @ w_out -> fp32. One wave per 16x16 tile.
// ---------------------------------------------------------------------------
__global__ __launch_bounds__(128) void k3_out(const f16* __restrict__ O,
                                              const f16* __restrict__ woutT,
                                              float* __restrict__ out) {
  const int wave = threadIdx.x >> 5;
  const int lane = threadIdx.x & 31;
  const int hi = lane >> 4;
  const int c = lane & 15;
  const int mt = blockIdx.x;
  const int nt = blockIdx.y * 4 + wave;  // 0..63
  v8f acc = {};
  const f16* arow = O + (size_t)mt * 16 * INNER_;
  const f16* bp = woutT + (size_t)nt * 16 * INNER_;
  for (int kk = 0; kk < INNER_; kk += 32) {
    acc = wmma_f16(load_a(arow + kk, INNER_, lane),
                   load_bt(bp + kk, INNER_, lane), acc);
  }
#pragma unroll
  for (int r = 0; r < 8; ++r) {
    out[(size_t)(mt * 16 + r + 8 * hi) * DIM_ + nt * 16 + c] = acc[r];
  }
}

// ---------------------------------------------------------------------------
extern "C" void kernel_launch(void* const* d_in, const int* in_sizes, int n_in,
                              void* d_out, int out_size, void* d_ws,
                              size_t ws_size, hipStream_t stream) {
  (void)in_sizes;
  (void)n_in;
  (void)out_size;
  (void)ws_size;
  const float* x = (const float*)d_in[0];
  const float* rope = (const float*)d_in[1];
  const unsigned char* mask = (const unsigned char*)d_in[2];
  const float* ln_w = (const float*)d_in[3];
  const float* ln_b = (const float*)d_in[4];
  const float* wqkv = (const float*)d_in[5];
  const float* bqkv = (const float*)d_in[6];
  const float* wout = (const float*)d_in[7];
  float* out = (float*)d_out;

  char* ws = (char*)d_ws;
  size_t off = 0;
  f16* xn = (f16*)(ws + off);
  off += (size_t)B_ * N_ * DIM_ * sizeof(f16);  // 32 MB (reused as O later)
  f16* wqkvT = (f16*)(ws + off);
  off += (size_t)3 * INNER_ * DIM_ * sizeof(f16);
  f16* woutT = (f16*)(ws + off);
  off += (size_t)DIM_ * INNER_ * sizeof(f16);
  f16* q = (f16*)(ws + off);
  off += (size_t)B_ * H_ * N_ * DH_ * sizeof(f16);
  f16* k = (f16*)(ws + off);
  off += (size_t)B_ * H_ * N_ * DH_ * sizeof(f16);
  f16* vT = (f16*)(ws + off);
  off += (size_t)B_ * H_ * N_ * DH_ * sizeof(f16);
  f16* O = xn;  // xn is dead after k1_qkv

  k0_ln<<<B_ * N_, 256, 0, stream>>>(x, ln_w, ln_b, xn);
  k0_twqkv<<<(DIM_ * 3 * INNER_ + 255) / 256, 256, 0, stream>>>(wqkv, wqkvT);
  k0_twout<<<(INNER_ * DIM_ + 255) / 256, 256, 0, stream>>>(wout, woutT);

  dim3 g1(B_ * N_ / 16, 6);
  k1_qkv<<<g1, 128, 0, stream>>>(xn, wqkvT, bqkv, rope, q, k, vT);

  k2_attn<<<B_ * H_ * N_ / 64, 128, 0, stream>>>(q, k, vT, mask, O);

  dim3 g3(B_ * N_ / 16, 16);
  k3_out<<<g3, 128, 0, stream>>>(O, woutT, out);
}
